// EnhancedSparseFeatureAttention_86028194939116
// MI455X (gfx1250) — matrix operand order repaired and checked by reading, not compiled, not run on hardware
//
#include <hip/hip_runtime.h>
#include <cstdint>
#include <cstddef>

#define DEV __device__ __forceinline__

typedef __bf16 bf16_t;
typedef __bf16 v16bf __attribute__((ext_vector_type(16)));
typedef __bf16 v8bf  __attribute__((ext_vector_type(8)));
typedef float  v8f   __attribute__((ext_vector_type(8)));
typedef float  v4f   __attribute__((ext_vector_type(4)));

constexpr int Bn = 256, Fn = 128, Dn = 512, Hn = 8, HDn = 64;
constexpr int KVALn = 64;   // top-k keep count (F/2)

struct alignas(8) bf16x4 { bf16_t v[4]; };

// ---------------- WMMA fragment helpers (layouts per ISA 7.12.2) ------------
// A (16x32 bf16): lane lo=l&15 is row M=lo; halves j: K = 16*(j>>3) + 8*hi + (j&7)
DEV v16bf load_a_frag(const bf16_t* Abase, int lda, int lane) {
  int lo = lane & 15, hi = lane >> 4;
  const bf16_t* p = Abase + (size_t)lo * lda + 8 * hi;
  v8bf a0 = *(const v8bf*)(p);
  v8bf a1 = *(const v8bf*)(p + 16);
  v16bf r;
#pragma unroll
  for (int i = 0; i < 8; ++i) { r[i] = a0[i]; r[8 + i] = a1[i]; }
  return r;
}

// B (32x16 bf16): lane lo=l&15 is column N=lo; halves j: K = 16*(j>>3) + 8*hi + (j&7)
DEV v16bf load_b_frag(const bf16_t* Bbase, int ldb, int lane) {
  int lo = lane & 15, hi = lane >> 4;
  const bf16_t* p = Bbase + lo;
  v16bf r;
#pragma unroll
  for (int i = 0; i < 8; ++i) {
    r[i] = p[(size_t)(8 * hi + i) * ldb];
    r[8 + i] = p[(size_t)(16 + 8 * hi + i) * ldb];
  }
  return r;
}

DEV v8f wmma_bf16(v16bf a, v16bf b, v8f c) {
  return __builtin_amdgcn_wmma_f32_16x16x32_bf16(false, a, false, b, (short)0, c,
                                                 false, false);
}

// ---------------- conversion kernel -----------------------------------------
__global__ void cvt_f32_bf16(const float* __restrict__ src, bf16_t* __restrict__ dst,
                             int n) {
  int i = blockIdx.x * blockDim.x + threadIdx.x;
  if (i < n) dst[i] = (bf16_t)src[i];
}

// ---------------- generic bf16 GEMM: C[M,512] = A[M,512] @ W[512,512] + bias -
// MODE 0: bf16 normal store; MODE 1: bf16 per-batch transposed store (kT);
// MODE 2: f32 normal store.  Wave = 16x64 tile; block = 8 waves.
template <int MODE>
__global__ void gemm512(const bf16_t* __restrict__ A, const bf16_t* __restrict__ W,
                        const float* __restrict__ bias, void* __restrict__ outp) {
  int lane = threadIdx.x & 31;
  int wave = threadIdx.x >> 5;
  int gw   = blockIdx.x * 8 + wave;
  int mTile = gw >> 3;        // M/16 tiles, M = 32768
  int n64   = gw & 7;         // N/64 groups, N = 512
  int m0 = mTile * 16, n0 = n64 * 64;
  int lo = lane & 15, hi = lane >> 4;

  v8f acc[4];
#pragma unroll
  for (int t = 0; t < 4; ++t)
#pragma unroll
    for (int r = 0; r < 8; ++r) acc[t][r] = 0.0f;

  for (int k0 = 0; k0 < Dn; k0 += 32) {
    v16bf a = load_a_frag(A + (size_t)m0 * Dn + k0, Dn, lane);
#pragma unroll
    for (int t = 0; t < 4; ++t) {
      v16bf b = load_b_frag(W + (size_t)k0 * Dn + n0 + 16 * t, Dn, lane);
      acc[t] = wmma_bf16(a, b, acc[t]);
    }
  }

#pragma unroll
  for (int t = 0; t < 4; ++t) {
    int n = n0 + 16 * t + lo;           // C/D: N = lane&15 (constant per lane)
    float bv = bias[n];
    if constexpr (MODE == 0) {
      bf16_t* o = (bf16_t*)outp;
#pragma unroll
      for (int r = 0; r < 8; ++r)
        o[(size_t)(m0 + 8 * hi + r) * Dn + n] = (bf16_t)(acc[t][r] + bv);
    } else if constexpr (MODE == 1) {
      // kT[b][n][f], tiles never cross batch boundary (16 | F)
      int bb = m0 / Fn, f0 = m0 % Fn;
      v8bf d;
#pragma unroll
      for (int r = 0; r < 8; ++r) d[r] = (bf16_t)(acc[t][r] + bv);
      bf16_t* o = (bf16_t*)outp;
      *(v8bf*)(o + ((size_t)bb * Dn + n) * Fn + f0 + 8 * hi) = d;
    } else {
      float* o = (float*)outp;
#pragma unroll
      for (int r = 0; r < 8; ++r)
        o[(size_t)(m0 + 8 * hi + r) * Dn + n] = acc[t][r] + bv;
    }
  }
}

// ---------------- pooling: combined[b][0:512]=price, [512:1024]=vol ---------
__global__ void pool_kernel(const float* __restrict__ x, float* __restrict__ comb) {
  int i = blockIdx.x * blockDim.x + threadIdx.x;
  if (i >= Bn * 2 * Dn) return;
  int b = i >> 10, c = i & 1023;
  int f0 = (c < Dn) ? 0 : (Fn / 2);
  int ch = (c < Dn) ? c : c - Dn;
  const float* p = x + ((size_t)b * Fn + f0) * Dn + ch;
  float s = 0.0f;
  for (int f = 0; f < Fn / 2; ++f) s += p[(size_t)f * Dn];
  comb[i] = s * (2.0f / Fn);
}

// ---------------- gating MLP: pv[b] = sigmoid(silu(comb@W1+b1)@W2+b2) -------
__global__ void mlp_kernel(const float* __restrict__ comb, const float* __restrict__ W1,
                           const float* __restrict__ b1, const float* __restrict__ W2,
                           const float* __restrict__ b2, float* __restrict__ pv) {
  __shared__ float sc[2 * Dn];
  __shared__ float red[256];
  int b = blockIdx.x, t = threadIdx.x;
  for (int i = t; i < 2 * Dn; i += 256) sc[i] = comb[(size_t)b * 2 * Dn + i];
  __syncthreads();
  float partial = 0.0f;
  for (int i = t; i < Dn; i += 256) {
    float a = b1[i];
    for (int c = 0; c < 2 * Dn; ++c) a += sc[c] * W1[(size_t)c * Dn + i];
    float s = a / (1.0f + __expf(-a));  // silu
    partial += s * W2[i];
  }
  red[t] = partial;
  __syncthreads();
  for (int s = 128; s > 0; s >>= 1) {
    if (t < s) red[t] += red[t + s];
    __syncthreads();
  }
  if (t == 0) {
    float z = red[0] + b2[0];
    pv[b] = 1.0f / (1.0f + __expf(-z));
  }
}

// ---------------- fused attention: one block per (b,h), 8 waves -------------
__global__ void attn_kernel(const bf16_t* __restrict__ qb, const bf16_t* __restrict__ kT,
                            const bf16_t* __restrict__ vb, const float* __restrict__ corr,
                            const float* __restrict__ fimp, const float* __restrict__ pv,
                            bf16_t* __restrict__ ctx) {
  __shared__ float sS[Fn * Fn];  // 64 KB scores; attn (bf16) overlaid in place
  int bh = blockIdx.x;
  int b = bh >> 3, h = bh & 7;
  int lane = threadIdx.x & 31, wave = threadIdx.x >> 5;
  int lo = lane & 15, hi = lane >> 4;

  // ---- phase 1: scores = q @ kT / sqrt(HD) -> LDS ----
  {
    v8f acc[8];
#pragma unroll
    for (int t = 0; t < 8; ++t)
#pragma unroll
      for (int r = 0; r < 8; ++r) acc[t][r] = 0.0f;
    const bf16_t* qbase = qb + ((size_t)(b * Fn + wave * 16)) * Dn + h * HDn;
    const bf16_t* kbase = kT + ((size_t)b * Dn + h * HDn) * Fn;
    for (int kb = 0; kb < HDn; kb += 32) {
      v16bf a = load_a_frag(qbase + kb, Dn, lane);
#pragma unroll
      for (int t = 0; t < 8; ++t) {
        v16bf bf = load_b_frag(kbase + (size_t)kb * Fn + 16 * t, Fn, lane);
        acc[t] = wmma_bf16(a, bf, acc[t]);
      }
    }
    const float scale = 0.125f;  // 1/sqrt(64)
#pragma unroll
    for (int t = 0; t < 8; ++t)
#pragma unroll
      for (int r = 0; r < 8; ++r)
        sS[(wave * 16 + 8 * hi + r) * Fn + 16 * t + lo] = acc[t][r] * scale;
  }
  __syncthreads();

  // ---- phase 2: priors + exact top-64 threshold + softmax (wave per row) ----
  {
    float pvb = pv[b];
    v4f fk = *(const v4f*)(fimp + lane * 4);
    for (int i = 0; i < 16; ++i) {
      int m = wave * 16 + i;
      v4f val = *(v4f*)(&sS[m * Fn + lane * 4]);
      v4f cp  = *(const v4f*)(corr + (size_t)m * Fn + lane * 4);
      float fq = fimp[m];
      bool mq = m < (Fn / 2);
      float vv[4];
      unsigned key[4];
#pragma unroll
      for (int j = 0; j < 4; ++j) {
        int col = lane * 4 + j;
        float v = val[j] + cp[j] * fq * fk[j];
        if ((col < (Fn / 2)) != mq) v += 0.5f * pvb;
        vv[j] = v;
        unsigned u = __float_as_uint(v);
        key[j] = (u & 0x80000000u) ? ~u : (u | 0x80000000u);  // monotonic key
      }
      // bisection for largest x with count_ge(x) >= KVAL  (== 64th-largest key)
      unsigned lok = 0u, hik = 0xFFFFFFFFu;
      while (lok < hik) {
        unsigned mid = (unsigned)(((unsigned long long)lok + hik + 1ull) >> 1);
        int c = (key[0] >= mid) + (key[1] >= mid) + (key[2] >= mid) + (key[3] >= mid);
        for (int s = 16; s > 0; s >>= 1) c += __shfl_xor(c, s, 32);
        if (c >= KVALn) lok = mid; else hik = mid - 1;
      }
      unsigned tk = lok;  // keep iff key >= tk  (== scores >= thresh, ties kept)
      float rmax = -1e30f;
#pragma unroll
      for (int j = 0; j < 4; ++j)
        if (key[j] >= tk) rmax = fmaxf(rmax, vv[j]);
      for (int s = 16; s > 0; s >>= 1) rmax = fmaxf(rmax, __shfl_xor(rmax, s, 32));
      float e[4], rsum = 0.0f;
#pragma unroll
      for (int j = 0; j < 4; ++j) {
        e[j] = (key[j] >= tk) ? __expf(vv[j] - rmax) : 0.0f;
        rsum += e[j];
      }
      for (int s = 16; s > 0; s >>= 1) rsum += __shfl_xor(rsum, s, 32);
      float inv = 1.0f / rsum;
      bf16_t* aRow = ((bf16_t*)sS) + (size_t)m * 256;  // bf16 overlay, stride 256
      bf16x4 ob;
#pragma unroll
      for (int j = 0; j < 4; ++j) ob.v[j] = (bf16_t)(e[j] * inv);
      *(bf16x4*)(aRow + lane * 4) = ob;  // in-place: all row reads already done
    }
  }
  __syncthreads();

  // ---- phase 3: ctx = attn @ v  (A from LDS bf16, B from global) ----
  {
    v8f acc[4];
#pragma unroll
    for (int t = 0; t < 4; ++t)
#pragma unroll
      for (int r = 0; r < 8; ++r) acc[t][r] = 0.0f;
    const bf16_t* aLds = (const bf16_t*)sS;
    for (int kb = 0; kb < Fn; kb += 32) {
      v16bf a = load_a_frag(aLds + (size_t)(wave * 16) * 256 + kb, 256, lane);
      const bf16_t* vbase = vb + ((size_t)(b * Fn + kb)) * Dn + h * HDn;
#pragma unroll
      for (int t = 0; t < 4; ++t) {
        v16bf bf = load_b_frag(vbase + 16 * t, Dn, lane);
        acc[t] = wmma_bf16(a, bf, acc[t]);
      }
    }
#pragma unroll
    for (int t = 0; t < 4; ++t)
#pragma unroll
      for (int r = 0; r < 8; ++r)
        ctx[((size_t)(b * Fn + wave * 16 + 8 * hi + r)) * Dn + h * HDn + 16 * t + lo] =
            (bf16_t)acc[t][r];
  }
}

// ---------------- host orchestration ----------------------------------------
extern "C" void kernel_launch(void* const* d_in, const int* in_sizes, int n_in,
                              void* d_out, int out_size, void* d_ws, size_t ws_size,
                              hipStream_t stream) {
  (void)in_sizes; (void)n_in; (void)out_size; (void)ws_size;
  const float* x    = (const float*)d_in[0];
  const float* Wq   = (const float*)d_in[1];
  const float* bq   = (const float*)d_in[2];
  const float* Wk   = (const float*)d_in[3];
  const float* bk   = (const float*)d_in[4];
  const float* Wv   = (const float*)d_in[5];
  const float* bv   = (const float*)d_in[6];
  const float* Wo   = (const float*)d_in[7];
  const float* bo   = (const float*)d_in[8];
  const float* corr = (const float*)d_in[9];
  const float* fimp = (const float*)d_in[10];
  const float* W1   = (const float*)d_in[11];
  const float* b1   = (const float*)d_in[12];
  const float* W2   = (const float*)d_in[13];
  const float* b2   = (const float*)d_in[14];
  float* out = (float*)d_out;

  char* ws = (char*)d_ws;
  size_t off = 0;
  auto alloc = [&](size_t bytes) {
    void* p = ws + off;
    off += (bytes + 255) & ~(size_t)255;
    return p;
  };
  bf16_t* xb   = (bf16_t*)alloc((size_t)Bn * Fn * Dn * 2);
  bf16_t* Wqb  = (bf16_t*)alloc((size_t)Dn * Dn * 2);
  bf16_t* Wkb  = (bf16_t*)alloc((size_t)Dn * Dn * 2);
  bf16_t* Wvb  = (bf16_t*)alloc((size_t)Dn * Dn * 2);
  bf16_t* Wob  = (bf16_t*)alloc((size_t)Dn * Dn * 2);
  bf16_t* qbp  = (bf16_t*)alloc((size_t)Bn * Fn * Dn * 2);
  bf16_t* kTb  = (bf16_t*)alloc((size_t)Bn * Dn * Fn * 2);
  bf16_t* vbp  = (bf16_t*)alloc((size_t)Bn * Fn * Dn * 2);
  bf16_t* ctx  = (bf16_t*)alloc((size_t)Bn * Fn * Dn * 2);
  float*  comb = (float*)alloc((size_t)Bn * 2 * Dn * 4);
  float*  pvb  = (float*)alloc((size_t)Bn * 4);

  const int nX = Bn * Fn * Dn, nW = Dn * Dn;
  cvt_f32_bf16<<<(nX + 255) / 256, 256, 0, stream>>>(x, xb, nX);
  cvt_f32_bf16<<<(nW + 255) / 256, 256, 0, stream>>>(Wq, Wqb, nW);
  cvt_f32_bf16<<<(nW + 255) / 256, 256, 0, stream>>>(Wk, Wkb, nW);
  cvt_f32_bf16<<<(nW + 255) / 256, 256, 0, stream>>>(Wv, Wvb, nW);
  cvt_f32_bf16<<<(nW + 255) / 256, 256, 0, stream>>>(Wo, Wob, nW);

  pool_kernel<<<(Bn * 2 * Dn + 255) / 256, 256, 0, stream>>>(x, comb);
  mlp_kernel<<<Bn, 256, 0, stream>>>(comb, W1, b1, W2, b2, pvb);

  const int gemmBlocks = (Bn * Fn / 16) * (Dn / 64) / 8;  // 2048
  gemm512<0><<<gemmBlocks, 256, 0, stream>>>(xb, Wqb, bq, qbp);
  gemm512<1><<<gemmBlocks, 256, 0, stream>>>(xb, Wkb, bk, kTb);
  gemm512<0><<<gemmBlocks, 256, 0, stream>>>(xb, Wvb, bv, vbp);

  attn_kernel<<<Bn * Hn, 256, 0, stream>>>(qbp, kTb, vbp, corr, fimp, pvb, ctx);

  gemm512<2><<<gemmBlocks, 256, 0, stream>>>(ctx, Wob, bo, out);
}
